// PyGCompatibleGCN_61864708932307
// MI455X (gfx1250) — matrix operand order repaired and checked by reading, not compiled, not run on hardware
//
#include <hip/hip_runtime.h>

// ---------------------------------------------------------------------------
// 2-layer GCN (PyG GCNConv semantics) for MI455X / gfx1250, wave32 + WMMA.
//   h1  = x @ W1            (bf16 WMMA, f32 accum)
//   a1  = scatter-add of dinv[s]*dinv[d] * h1[s]  (incl. self loops) + b1, relu
//   h2  = a1 @ W2           (bf16 WMMA, f32 accum)
//   a2  = scatter-add ... + b2
//   out = log_softmax(a2, axis=1)
// Aggregations are L2-resident on MI455X (h1/acc each 51.2 MB < 192 MB L2).
// ---------------------------------------------------------------------------

typedef __attribute__((ext_vector_type(16))) __bf16 v16bf;
typedef __attribute__((ext_vector_type(8)))  __bf16 v8bf;
typedef __attribute__((ext_vector_type(4)))  __bf16 v4bf;
typedef __attribute__((ext_vector_type(8)))  float  v8f;

union FragBF { v16bf v; v8bf h[2]; };

// ---- f32 -> bf16 bulk convert (4 elems/thread, vectorized) ----------------
__global__ void k_f32_to_bf16(const float* __restrict__ in,
                              __bf16* __restrict__ out, long n4) {
  long i = (long)blockIdx.x * blockDim.x + threadIdx.x;
  if (i >= n4) return;
  float4 v = ((const float4*)in)[i];
  v4bf o;
  o[0] = (__bf16)v.x; o[1] = (__bf16)v.y; o[2] = (__bf16)v.z; o[3] = (__bf16)v.w;
  ((v4bf*)out)[i] = o;
}

// ---- pack W[K][Ncol] (row-major f32) into WMMA B-fragment order (bf16) ----
// Fragment element i of lane L of tile (nt, kt) is
//   B[kt*32 + (L>>4)*16 + i][nt*16 + (L&15)]
// stored at Wp[((nt*kTiles + kt)*32 + L)*16 + i]  -> GEMM loads are 2x b128.
__global__ void k_pack_w(const float* __restrict__ W, __bf16* __restrict__ Wp,
                         int K, int Ncol) {
  int idx = blockIdx.x * blockDim.x + threadIdx.x;
  if (idx >= K * Ncol) return;
  int i    = idx & 15;
  int lane = (idx >> 4) & 31;
  int t    = idx >> 9;
  int kTiles = K >> 5;
  int kt = t % kTiles;
  int nt = t / kTiles;
  int n = nt * 16 + (lane & 15);
  int k = kt * 32 + (lane >> 4) * 16 + i;
  Wp[idx] = (__bf16)W[k * Ncol + n];
}

// ---- constant fill --------------------------------------------------------
__global__ void k_fill(float* __restrict__ p, float v, long n) {
  long i = (long)blockIdx.x * blockDim.x + threadIdx.x;
  if (i < n) p[i] = v;
}

// ---- degree accumulation over targets (deg pre-initialized to 1 = loop) ---
__global__ void k_deg(const int* __restrict__ dst, float* __restrict__ deg, int E) {
  int e = blockIdx.x * blockDim.x + threadIdx.x;
  if (e < E) atomicAdd(&deg[dst[e]], 1.0f);
}

// ---- deg -> deg^{-1/2} in place ------------------------------------------
__global__ void k_rsqrt(float* __restrict__ d, int n) {
  int i = blockIdx.x * blockDim.x + threadIdx.x;
  if (i < n) { float v = d[i]; d[i] = v > 0.0f ? rsqrtf(v) : 0.0f; }
}

// ---- bf16 WMMA GEMM: C[M][Ncol] = A[M][K] @ B, f32 accumulate -------------
// One wave per 16x16 tile; 4 waves/block cover 64 columns; loop over K.
__global__ void __launch_bounds__(128)
k_gemm_bf16_wmma(const __bf16* __restrict__ A, const __bf16* __restrict__ Bp,
                 float* __restrict__ C, int M, int K, int Ncol) {
  const int lane  = threadIdx.x & 31;
  const int wave  = threadIdx.x >> 5;
  const int m0    = blockIdx.x * 16;
  const int ntile = blockIdx.y * 4 + wave;           // 16-column tile index
  if (ntile * 16 >= Ncol) return;                    // wave-uniform exit
  const int n0 = ntile * 16;
  const int mh = lane & 15;                          // row (A) / col (C) in tile
  const int hi = lane >> 4;
  const int kTiles = K >> 5;

  const __bf16* arow  = A  + (long)(m0 + mh) * K + hi * 16;
  const __bf16* bbase = Bp + (long)ntile * kTiles * 512 + lane * 16;

  v8f c = {};
  for (int kt = 0; kt < kTiles; ++kt) {
    FragBF a, b;
    const __bf16* ap = arow + kt * 32;
    a.h[0] = *(const v8bf*)(ap);
    a.h[1] = *(const v8bf*)(ap + 8);
    const __bf16* bp = bbase + (long)kt * 512;
    b.h[0] = *(const v8bf*)(bp);
    b.h[1] = *(const v8bf*)(bp + 8);
    c = __builtin_amdgcn_wmma_f32_16x16x32_bf16(
        /*neg_a=*/false, a.v, /*neg_b=*/false, b.v,
        /*c_mod=*/(short)0, c, /*reuse_a=*/false, /*reuse_b=*/false);
  }
#pragma unroll
  for (int r = 0; r < 8; ++r)
    C[(long)(m0 + hi * 8 + r) * Ncol + n0 + mh] = c[r];
}

// ---- normalized scatter-add aggregation (E real edges + N self loops) -----
// thread = (edge, 4-feature group); gathers float4 from h[s], atomic-adds
// into out[d]. Both arrays are L2-resident on MI455X.
__global__ void k_aggregate(const float* __restrict__ h,
                            const float* __restrict__ dinv,
                            const int* __restrict__ src,
                            const int* __restrict__ dst,
                            float* __restrict__ out, int E, int N, int F) {
  const int  q   = F >> 2;
  long tid = (long)blockIdx.x * blockDim.x + threadIdx.x;
  long e   = tid / q;
  if (e >= (long)E + N) return;
  int f = (int)(tid % q) * 4;
  int s, d;
  if (e < E) { s = src[e]; d = dst[e]; }
  else       { s = d = (int)(e - E); }               // self loop
  float nrm = dinv[s] * dinv[d];
  const float4 v = *(const float4*)(h + (long)s * F + f);
  float* o = out + (long)d * F + f;
  atomicAdd(o + 0, v.x * nrm);
  atomicAdd(o + 1, v.y * nrm);
  atomicAdd(o + 2, v.z * nrm);
  atomicAdd(o + 3, v.w * nrm);
}

// ---- (agg + bias) -> relu -> bf16 (feeds GEMM2) ---------------------------
__global__ void k_bias_relu_bf16(const float* __restrict__ in,
                                 const float* __restrict__ bias,
                                 __bf16* __restrict__ out, long n, int F) {
  long i = (long)blockIdx.x * blockDim.x + threadIdx.x;
  if (i >= n) return;
  int f = (int)(i % F);
  float v = in[i] + bias[f];
  out[i] = (__bf16)fmaxf(v, 0.0f);
}

// ---- per-row (64 feats) bias + log_softmax; one wave32 per row ------------
__global__ void __launch_bounds__(256)
k_logsoftmax64(const float* __restrict__ in, const float* __restrict__ bias,
               float* __restrict__ out, int N) {
  int row  = blockIdx.x * 8 + (threadIdx.x >> 5);
  int lane = threadIdx.x & 31;
  if (row >= N) return;
  const float2 v  = *(const float2*)(in + (long)row * 64 + lane * 2);
  const float2 bb = *(const float2*)(bias + lane * 2);
  float a = v.x + bb.x;
  float b = v.y + bb.y;
  float mx = fmaxf(a, b);
#pragma unroll
  for (int off = 16; off > 0; off >>= 1)
    mx = fmaxf(mx, __shfl_xor(mx, off, 32));
  float s = expf(a - mx) + expf(b - mx);
#pragma unroll
  for (int off = 16; off > 0; off >>= 1)
    s += __shfl_xor(s, off, 32);
  float lse = mx + logf(s);
  float2 o; o.x = a - lse; o.y = b - lse;
  *(float2*)(out + (long)row * 64 + lane * 2) = o;
}

// ---------------------------------------------------------------------------
extern "C" void kernel_launch(void* const* d_in, const int* in_sizes, int n_in,
                              void* d_out, int out_size, void* d_ws, size_t ws_size,
                              hipStream_t stream) {
  const float* x  = (const float*)d_in[0];
  const int*   ei = (const int*)d_in[1];   // int32 under default JAX config
  const float* W1 = (const float*)d_in[2];
  const float* b1 = (const float*)d_in[3];
  const float* W2 = (const float*)d_in[4];
  const float* b2 = (const float*)d_in[5];

  const int Fin = 512, Fh = 256, Fout = 64;
  const int N = in_sizes[0] / Fin;         // 50000
  const int E = in_sizes[1] / 2;           // 800000
  const int* src = ei;
  const int* dst = ei + E;

  // ---- workspace layout (with reuse; ~128.5 MB total) ----
  char* w = (char*)d_ws;
  size_t off = 0;
  __bf16* xb   = (__bf16*)(w + off);                 // N*Fin bf16
  float*  agg1 = (float*)(w + off);                  // reuse same bytes after GEMM1
  off += (size_t)N * Fin * sizeof(__bf16);           // == N*Fh*4
  float* h1   = (float*)(w + off);                   // N*Fh f32
  float* h2   = (float*)(w + off);                   // reuse after relu-convert
  float* agg2 = (float*)(w + off + (size_t)N * Fout * sizeof(float));
  off += (size_t)N * Fh * sizeof(float);
  __bf16* h1b = (__bf16*)(w + off); off += (size_t)N * Fh * sizeof(__bf16);
  __bf16* w1p = (__bf16*)(w + off); off += (size_t)Fin * Fh * sizeof(__bf16);
  __bf16* w2p = (__bf16*)(w + off); off += (size_t)Fh * Fout * sizeof(__bf16);
  float*  dinv = (float*)(w + off); off += (size_t)N * sizeof(float);
  (void)ws_size; (void)n_in; (void)out_size;

  // 1) precision conversion / weight packing
  long nx4 = (long)N * Fin / 4;
  k_f32_to_bf16<<<(unsigned)((nx4 + 255) / 256), 256, 0, stream>>>(x, xb, nx4);
  k_pack_w<<<(unsigned)((Fin * Fh + 255) / 256), 256, 0, stream>>>(W1, w1p, Fin, Fh);
  k_pack_w<<<(unsigned)((Fh * Fout + 255) / 256), 256, 0, stream>>>(W2, w2p, Fh, Fout);

  // 2) symmetric normalization coefficients (deg over targets, self-loop +1)
  k_fill<<<(unsigned)((N + 255) / 256), 256, 0, stream>>>(dinv, 1.0f, (long)N);
  k_deg<<<(unsigned)((E + 255) / 256), 256, 0, stream>>>(dst, dinv, E);
  k_rsqrt<<<(unsigned)((N + 255) / 256), 256, 0, stream>>>(dinv, N);

  // 3) layer 1: GEMM -> aggregate -> bias+relu (+bf16 for GEMM2)
  dim3 g1((unsigned)(N / 16), (unsigned)(Fh / 64));
  k_gemm_bf16_wmma<<<g1, 128, 0, stream>>>(xb, w1p, h1, N, Fin, Fh);
  long nAgg1 = (long)N * Fh;
  k_fill<<<(unsigned)((nAgg1 + 255) / 256), 256, 0, stream>>>(agg1, 0.0f, nAgg1);
  long t1 = ((long)E + N) * (Fh / 4);
  k_aggregate<<<(unsigned)((t1 + 255) / 256), 256, 0, stream>>>(
      h1, dinv, src, dst, agg1, E, N, Fh);
  k_bias_relu_bf16<<<(unsigned)((nAgg1 + 255) / 256), 256, 0, stream>>>(
      agg1, b1, h1b, nAgg1, Fh);

  // 4) layer 2: GEMM -> aggregate -> bias + log_softmax
  dim3 g2((unsigned)(N / 16), (unsigned)(Fout / 64));
  k_gemm_bf16_wmma<<<g2, 128, 0, stream>>>(h1b, w2p, h2, N, Fh, Fout);
  long nAgg2 = (long)N * Fout;
  k_fill<<<(unsigned)((nAgg2 + 255) / 256), 256, 0, stream>>>(agg2, 0.0f, nAgg2);
  long t2 = ((long)E + N) * (Fout / 4);
  k_aggregate<<<(unsigned)((t2 + 255) / 256), 256, 0, stream>>>(
      h2, dinv, src, dst, agg2, E, N, Fout);
  k_logsoftmax64<<<(unsigned)((N + 7) / 8), 256, 0, stream>>>(
      agg2, b2, (float*)d_out, N);
}
